// MambaGenerator_16449724745406
// MI455X (gfx1250) — compile-verified, matrix operand order checked by reading, not run on hardware
//
#include <hip/hip_runtime.h>
#include <cstdint>

typedef __attribute__((ext_vector_type(16))) _Float16 v16h;
typedef __attribute__((ext_vector_type(8)))  _Float16 v8h;
typedef __attribute__((ext_vector_type(8)))  float    v8f;
typedef __attribute__((ext_vector_type(4)))  unsigned int u32x4;
typedef __attribute__((ext_vector_type(8)))  int      i32x8;
typedef __attribute__((ext_vector_type(4)))  int      i32x4;

#if defined(__gfx1250__) && __has_builtin(__builtin_amdgcn_tensor_load_to_lds) && __has_builtin(__builtin_amdgcn_s_wait_tensorcnt)
#define USE_TDM 1
#else
#define USE_TDM 0
#endif

// ---------------------------------------------------------------------------
// Tensor Data Mover: 2D f16 tile (tileK contiguous elems x tileRows rows) -> LDS
// D# packing per CDNA5 ISA (group0/group1).
// ---------------------------------------------------------------------------
static __device__ __forceinline__ void tdm_load_tile_f16(
    void* ldsDst, const _Float16* gSrc,
    unsigned tileK, unsigned tileRows,
    unsigned long long strideElems, unsigned dim0, unsigned dim1)
{
#if USE_TDM
  unsigned long long ga = (unsigned long long)(uintptr_t)gSrc;
  unsigned la = (unsigned)(uintptr_t)ldsDst;
  u32x4 g0;
  g0.x = 1u;                                                  // count=1 (valid D#)
  g0.y = la;                                                  // lds_addr [63:32]
  g0.z = (unsigned)(ga & 0xFFFFFFFFull);                      // global_addr lo
  g0.w = (unsigned)((ga >> 32) & 0x1FFFFFFull) | (2u << 30);  // addr hi + type=2
  i32x8 g1;
  g1[0] = (int)(1u << 16);                                    // wg_mask=0, data_size=1 (2B)
  g1[1] = (int)((dim0 & 0xFFFFu) << 16);                      // tensor_dim0[15:0] @ [63:48]
  g1[2] = (int)((dim0 >> 16) | ((dim1 & 0xFFFFu) << 16));     // dim0 hi | tensor_dim1 lo
  g1[3] = (int)((dim1 >> 16) | ((tileK & 0xFFFFu) << 16));    // dim1 hi | tile_dim0
  g1[4] = (int)(tileRows & 0xFFFFu);                          // tile_dim1 (tile_dim2=0)
  g1[5] = (int)(unsigned)(strideElems & 0xFFFFFFFFull);       // tensor_dim0_stride lo
  g1[6] = (int)(unsigned)((strideElems >> 32) & 0xFFFFull);   // stride hi (stride1=0)
  g1[7] = 0;
  i32x4 z4 = {0, 0, 0, 0};
#if __clang_major__ >= 23
  i32x8 z8 = {0, 0, 0, 0, 0, 0, 0, 0};
  __builtin_amdgcn_tensor_load_to_lds(g0, g1, z4, z4, z8, 0);
#else
  __builtin_amdgcn_tensor_load_to_lds(g0, g1, z4, z4, 0);
#endif
#else
  (void)ldsDst; (void)gSrc; (void)tileK; (void)tileRows;
  (void)strideElems; (void)dim0; (void)dim1;
#endif
}

// ---------------------------------------------------------------------------
// WMMA fragment load from LDS, ISA §7.12.2 16-bit A/B layout (ld = 32 elems):
// lanes 0-15 hold row (row0+lane): K {0..7,16..23}; lanes 16-31: K {8..15,24..31}
// ---------------------------------------------------------------------------
static __device__ __forceinline__ v16h load_frag(const _Float16* base, int row0,
                                                 int lane) {
  const _Float16* p = base + (size_t)(row0 + (lane & 15)) * 32 + ((lane >> 4) << 3);
  v8h lo = *(const v8h*)p;
  v8h hi = *(const v8h*)(p + 16);
  v16h r;
#pragma unroll
  for (int i = 0; i < 8; ++i) { r[i] = lo[i]; r[8 + i] = hi[i]; }
  return r;
}

// ---------------------------------------------------------------------------
// WMMA GEMM: C[M,N] = A[M,K] * W[N,K]^T  (both operands K-major, f16 in / f32 acc)
// Block: 256 threads = 8 waves (4 along M x 2 along N). Block tile 128 x TN,
// wave tile 32 x (TN/2), K-step 32. Both tiles staged by the Tensor Data Mover
// into double-buffered LDS; next tile's TDM issue overlaps current compute.
// mode 0: store f32 C and/or f16 C.  mode 1: transposed store into d_out + bias
//         (row m = b*512 + l  ->  out[b, n, tOff + l],  out is [8,128,4096]).
// ---------------------------------------------------------------------------
#define GEMM_TM 128
#define GEMM_TK 32

template <int TN>
__global__ __launch_bounds__(256) void gemm_wmma(
    const _Float16* __restrict__ A, int lda,
    const _Float16* __restrict__ W, int ldw,
    int Mtot, int Ntot, int Ktot,
    float* __restrict__ Cf32, _Float16* __restrict__ Cf16, int ldc,
    int mode, const float* __restrict__ bias, float* __restrict__ outT, int tOff)
{
  constexpr int NB = TN / 32;          // 16-col WMMA tiles per wave (N) = NB per i
  __shared__ __align__(16) _Float16 As[2][GEMM_TM * GEMM_TK];
  __shared__ __align__(16) _Float16 Ws[2][TN * GEMM_TK];

  const int tid  = threadIdx.x;
  const int lane = tid & 31;
  const int wid  = tid >> 5;
  const int mTile = blockIdx.x * GEMM_TM;
  const int nTile = blockIdx.y * TN;
  const int wm = (wid & 3) * 32;                // wave row offset in block tile
  const int wn = (wid >> 2) * (TN / 2);         // wave col offset in block tile

  v8f acc[2][NB];
#pragma unroll
  for (int i = 0; i < 2; ++i)
#pragma unroll
    for (int j = 0; j < NB; ++j) acc[i][j] = v8f{};

  const int nk = Ktot / GEMM_TK;

#if USE_TDM
  if (wid == 0) {   // prologue: stage tile 0 into buffer 0 (2 TDM descriptors)
    tdm_load_tile_f16(&As[0][0], A + (size_t)mTile * lda, GEMM_TK, GEMM_TM,
                      (unsigned long long)lda, (unsigned)Ktot, (unsigned)Mtot);
    tdm_load_tile_f16(&Ws[0][0], W + (size_t)nTile * ldw, GEMM_TK, TN,
                      (unsigned long long)ldw, (unsigned)Ktot, (unsigned)Ntot);
  }
#endif

  for (int k = 0; k < nk; ++k) {
    const int cur = k & 1;
    __syncthreads();                  // buf[cur^1] free: tile k-1 fully consumed
#if USE_TDM
    if (wid == 0) {
      if (k + 1 < nk) {               // overlap: issue tile k+1, wait only tile k
        const int kk = (k + 1) * GEMM_TK;
        tdm_load_tile_f16(&As[cur ^ 1][0], A + (size_t)mTile * lda + kk,
                          GEMM_TK, GEMM_TM, (unsigned long long)lda,
                          (unsigned)Ktot, (unsigned)Mtot);
        tdm_load_tile_f16(&Ws[cur ^ 1][0], W + (size_t)nTile * ldw + kk,
                          GEMM_TK, TN, (unsigned long long)ldw,
                          (unsigned)Ktot, (unsigned)Ntot);
        __builtin_amdgcn_s_wait_tensorcnt(2);   // tile k's pair retired
      } else {
        __builtin_amdgcn_s_wait_tensorcnt(0);   // last tile: drain
      }
    }
#else
    {
      const int kk = k * GEMM_TK;
      for (int c = tid; c < (GEMM_TM * GEMM_TK) / 8; c += 256) {
        int row = c >> 2, seg = c & 3;
        *(v8h*)(&As[cur][row * GEMM_TK + seg * 8]) =
            *(const v8h*)(A + (size_t)(mTile + row) * lda + kk + seg * 8);
      }
      for (int c = tid; c < (TN * GEMM_TK) / 8; c += 256) {
        int row = c >> 2, seg = c & 3;
        *(v8h*)(&Ws[cur][row * GEMM_TK + seg * 8]) =
            *(const v8h*)(W + (size_t)(nTile + row) * ldw + kk + seg * 8);
      }
    }
#endif
    __syncthreads();                  // buf[cur] visible to all waves

    v16h a0 = load_frag(&As[cur][0], wm,      lane);
    v16h a1 = load_frag(&As[cur][0], wm + 16, lane);
#pragma unroll
    for (int j = 0; j < NB; ++j) {
      v16h b = load_frag(&Ws[cur][0], wn + j * 16, lane);
      acc[0][j] = __builtin_amdgcn_wmma_f32_16x16x32_f16(false, a0, false, b,
                                                         (short)0, acc[0][j], false, false);
      acc[1][j] = __builtin_amdgcn_wmma_f32_16x16x32_f16(false, a1, false, b,
                                                         (short)0, acc[1][j], false, false);
    }
  }

  // C/D layout: lanes 0-15 -> N=lane, M=vgpr; lanes 16-31 -> N=lane-16, M=vgpr+8
  const int mo = (lane >> 4) << 3;
  const int nc = lane & 15;
#pragma unroll
  for (int i = 0; i < 2; ++i)
#pragma unroll
    for (int j = 0; j < NB; ++j) {
#pragma unroll
      for (int r = 0; r < 8; ++r) {
        int m = mTile + wm + i * 16 + mo + r;
        int n = nTile + wn + j * 16 + nc;
        float v = acc[i][j][r];
        if (mode == 0) {
          if (Cf32) Cf32[(size_t)m * ldc + n] = v;
          if (Cf16) Cf16[(size_t)m * ldc + n] = (_Float16)v;
        } else {
          int b = m >> 9, l = m & 511;
          outT[((size_t)b * 128 + n) * 4096 + tOff + l] = v + bias[n];
        }
      }
    }
}

// ---------------------------------------------------------------------------
// Elementwise / small kernels
// ---------------------------------------------------------------------------
__global__ void cvt_f32_f16(const float* __restrict__ s, _Float16* __restrict__ d, int n) {
  int i = blockIdx.x * blockDim.x + threadIdx.x;
  if (i < n) d[i] = (_Float16)s[i];
}

__global__ void aneg_kernel(const float* __restrict__ Al, float* __restrict__ An, int n) {
  int i = blockIdx.x * blockDim.x + threadIdx.x;
  if (i < n) An[i] = -__expf(Al[i]);
}

// depthwise conv (K=4, causal) + bias + SiLU over (B,L,D_INNER); xraw = xz[..., :1024]
__global__ __launch_bounds__(256) void conv_silu_kernel(
    const float* __restrict__ xz, const float* __restrict__ cw,
    const float* __restrict__ cb, float* __restrict__ xcf, _Float16* __restrict__ xch)
{
  int idx = blockIdx.x * 256 + threadIdx.x;   // over 4096*1024
  int e  = idx & 1023;
  int ml = idx >> 10;
  int l  = ml & 511;
  int bl0 = ml - l;                            // b*512
  float acc = cb[e];
#pragma unroll
  for (int k = 0; k < 4; ++k) {
    int lp = l - 3 + k;
    if (lp >= 0) acc += cw[e * 4 + k] * xz[(size_t)(bl0 + lp) * 2048 + e];
  }
  float s = acc / (1.f + __expf(-acc));        // SiLU
  xcf[idx] = s;
  xch[idx] = (_Float16)s;
}

__global__ __launch_bounds__(256) void softplus_kernel(
    float* __restrict__ dtb, const float* __restrict__ b_dt)
{
  int idx = blockIdx.x * 256 + threadIdx.x;
  int e = idx & 1023;
  float v = dtb[idx] + b_dt[e];
  dtb[idx] = (v > 20.f) ? v : log1pf(__expf(v));
}

__global__ __launch_bounds__(256) void gate_kernel(
    const float* __restrict__ ys, const float* __restrict__ xcf,
    const float* __restrict__ xz, const float* __restrict__ Dv,
    _Float16* __restrict__ ygh)
{
  int idx = blockIdx.x * 256 + threadIdx.x;
  int e  = idx & 1023;
  int ml = idx >> 10;
  float z  = xz[(size_t)ml * 2048 + 1024 + e];
  float sz = z / (1.f + __expf(-z));
  ygh[idx] = (_Float16)((ys[idx] + Dv[e] * xcf[idx]) * sz);
}

// Selective scan: block = one batch, thread = one channel e; h[16] in registers,
// B/C (32 floats per timestep) broadcast via LDS.
__global__ __launch_bounds__(1024) void scan_kernel(
    const float* __restrict__ xdbl, const float* __restrict__ dtb,
    const float* __restrict__ xc, const float* __restrict__ Aneg,
    float* __restrict__ y)
{
  __shared__ float sBC[32];
  const int b = blockIdx.x;
  const int e = threadIdx.x;
  float a[16], h[16];
#pragma unroll
  for (int s = 0; s < 16; ++s) { a[s] = Aneg[e * 16 + s]; h[s] = 0.f; }
  for (int l = 0; l < 512; ++l) {
    size_t row = (size_t)(b * 512 + l);
    if (e < 32) sBC[e] = xdbl[row * 64 + 32 + e];
    __syncthreads();
    float dt  = dtb[row * 1024 + e];
    float x   = xc [row * 1024 + e];
    float dtx = dt * x;
    float acc = 0.f;
#pragma unroll
    for (int s = 0; s < 16; ++s) {
      h[s] = h[s] * __expf(dt * a[s]) + dtx * sBC[s];
      acc += h[s] * sBC[16 + s];
    }
    y[row * 1024 + e] = acc;
    __syncthreads();
  }
}

// ---------------------------------------------------------------------------
// Host orchestration
// ---------------------------------------------------------------------------
extern "C" void kernel_launch(void* const* d_in, const int* in_sizes, int n_in,
                              void* d_out, int out_size, void* d_ws, size_t ws_size,
                              hipStream_t stream) {
  (void)in_sizes; (void)n_in; (void)out_size; (void)ws_size;
  const float* x      = (const float*)d_in[0];
  const float* W_in   = (const float*)d_in[1];
  const float* conv_w = (const float*)d_in[2];
  const float* conv_b = (const float*)d_in[3];
  const float* W_x    = (const float*)d_in[4];
  const float* W_dt   = (const float*)d_in[5];
  const float* b_dt   = (const float*)d_in[6];
  const float* A_log  = (const float*)d_in[7];
  const float* Dvec   = (const float*)d_in[8];
  const float* W_out  = (const float*)d_in[9];
  const float* W_proj = (const float*)d_in[10];
  const float* b_proj = (const float*)d_in[11];
  float* out = (float*)d_out;

  const int M = 4096;  // B*L

  char* p = (char*)d_ws;
  auto alloc = [&](size_t bytes) -> char* {
    char* r = p; p += (bytes + 255) & ~size_t(255); return r;
  };
  _Float16* W_in_h   = (_Float16*)alloc((size_t)2048 * 512 * 2);
  _Float16* W_x_h    = (_Float16*)alloc((size_t)64   * 1024 * 2);
  _Float16* W_dt_h   = (_Float16*)alloc((size_t)1024 * 32 * 2);
  _Float16* W_out_h  = (_Float16*)alloc((size_t)512  * 1024 * 2);
  _Float16* W_proj_h = (_Float16*)alloc((size_t)128  * 512 * 2);
  float*    Aneg     = (float*)   alloc((size_t)1024 * 16 * 4);
  _Float16* u_h      = (_Float16*)alloc((size_t)M * 512 * 2);
  float*    xz       = (float*)   alloc((size_t)M * 2048 * 4);
  float*    xc_f     = (float*)   alloc((size_t)M * 1024 * 4);
  _Float16* xc_h     = (_Float16*)alloc((size_t)M * 1024 * 2);
  float*    xdbl_f   = (float*)   alloc((size_t)M * 64 * 4);
  _Float16* xdbl_h   = (_Float16*)alloc((size_t)M * 64 * 2);
  float*    dtb      = (float*)   alloc((size_t)M * 1024 * 4);
  float*    yscan    = (float*)   alloc((size_t)M * 1024 * 4);
  _Float16* yg_h     = (_Float16*)alloc((size_t)M * 1024 * 2);

  // one-time weight conversion + A precompute + input conversion
  cvt_f32_f16<<<dim3(4096), 256, 0, stream>>>(W_in,   W_in_h,   2048 * 512);
  cvt_f32_f16<<<dim3(256),  256, 0, stream>>>(W_x,    W_x_h,    64 * 1024);
  cvt_f32_f16<<<dim3(128),  256, 0, stream>>>(W_dt,   W_dt_h,   1024 * 32);
  cvt_f32_f16<<<dim3(2048), 256, 0, stream>>>(W_out,  W_out_h,  512 * 1024);
  cvt_f32_f16<<<dim3(256),  256, 0, stream>>>(W_proj, W_proj_h, 128 * 512);
  aneg_kernel<<<dim3(64),   256, 0, stream>>>(A_log,  Aneg,     1024 * 16);
  cvt_f32_f16<<<dim3(8192), 256, 0, stream>>>(x,      u_h,      M * 512);

  for (int t = 0; t < 8; ++t) {
    // xz = u @ W_in^T      (4096 x 2048 x 512)
    gemm_wmma<128><<<dim3(32, 16), 256, 0, stream>>>(u_h, 512, W_in_h, 512,
        M, 2048, 512, xz, nullptr, 2048, 0, nullptr, nullptr, 0);
    // causal depthwise conv + SiLU
    conv_silu_kernel<<<dim3(16384), 256, 0, stream>>>(xz, conv_w, conv_b, xc_f, xc_h);
    // x_dbl = xc @ W_x^T   (4096 x 64 x 1024)
    gemm_wmma<64><<<dim3(32, 1), 256, 0, stream>>>(xc_h, 1024, W_x_h, 1024,
        M, 64, 1024, xdbl_f, xdbl_h, 64, 0, nullptr, nullptr, 0);
    // dt_lin = dt_r @ W_dt^T  (4096 x 1024 x 32); dt_r = x_dbl[:, :32]
    gemm_wmma<128><<<dim3(32, 8), 256, 0, stream>>>(xdbl_h, 64, W_dt_h, 32,
        M, 1024, 32, dtb, nullptr, 1024, 0, nullptr, nullptr, 0);
    // dt = softplus(dt_lin + b_dt), in place
    softplus_kernel<<<dim3(16384), 256, 0, stream>>>(dtb, b_dt);
    // selective scan over L
    scan_kernel<<<dim3(8), 1024, 0, stream>>>(xdbl_f, dtb, xc_f, Aneg, yscan);
    // y = (y + D*xc) * silu(z)
    gate_kernel<<<dim3(16384), 256, 0, stream>>>(yscan, xc_f, xz, Dvec, yg_h);
    // next input u = y @ W_out^T (4096 x 512 x 1024), f16 only
    gemm_wmma<128><<<dim3(32, 4), 256, 0, stream>>>(yg_h, 1024, W_out_h, 1024,
        M, 512, 1024, nullptr, u_h, 512, 0, nullptr, nullptr, 0);
    // out[:, :, t*512:(t+1)*512] = (u @ W_proj^T + b_proj), transposed store
    gemm_wmma<128><<<dim3(32, 1), 256, 0, stream>>>(u_h, 512, W_proj_h, 512,
        M, 128, 512, nullptr, nullptr, 0, 1, b_proj, out, t * 512);
  }
}
